// UnifiedRadialMLP_64544768524905
// MI455X (gfx1250) — matrix operand order repaired and checked by reading, not compiled, not run on hardware
//
#include <hip/hip_runtime.h>
#include <math.h>

#define N_EXP 8
#define D_IN  64
#define H     128
#define D_OUT 64
#define TILE_E 64
#define XS 72    // Xt row stride (bf16 elems)
#define AS 136   // Ab row stride (bf16 elems)
#define CS 132   // Acc row stride (f32 elems)

#define FRAG_ELEMS 512               // one 32x16 B-fragment: 32 lanes x 16 bf16
#define FRAGS_PER_EXPERT 64          // L1:16  L2:32  L3:16
#define WS_NEEDED ((size_t)N_EXP * FRAGS_PER_EXPERT * FRAG_ELEMS * 2)  // 512 KB

typedef __attribute__((ext_vector_type(16))) __bf16 v16bf;
typedef __attribute__((ext_vector_type(2)))  __bf16 v2bf;
typedef __attribute__((ext_vector_type(8)))  float  v8f;

// ---------------------------------------------------------------------------
// A operand: 16x32 bf16 tile from LDS (row-major, given stride).
// ISA layout: lanes 0-15 -> row m0+lane, K = k0+0..7 and k0+16..23
//             lanes 16-31 -> row m0+lane-16, K = k0+8..15 and k0+24..31
__device__ __forceinline__ v16bf load_a_lds(const __bf16* A, int stride,
                                            int m0, int k0, int lane) {
  const int hi = lane >> 4;
  const int m  = m0 + (lane & 15);
  const __bf16* p = A + m * stride + k0 + hi * 8;
  v16bf a;
#pragma unroll
  for (int i = 0; i < 8; ++i) { a[i] = p[i]; a[i + 8] = p[16 + i]; }
  return a;
}

// B operand, packed path: fragment pre-transposed to bf16 by pack_weights.
// Per lane: 32 contiguous bytes -> two global_load_b128, no conversion VALU.
__device__ __forceinline__ v16bf load_frag(const __bf16* __restrict__ Wp,
                                           int frag, int lane) {
  return *(const v16bf*)(Wp + (size_t)frag * FRAG_ELEMS + lane * 16);
}

// B operand, fallback path: gather transpose of W[n0..n0+15][k0..k0+31]
// (row-major [Nout][K] f32) with native cvt. Lanes index N, elements index K (same K order as A).
template<int K>
__device__ __forceinline__ v16bf load_b_gather(const float* __restrict__ W,
                                               int n0, int k0, int lane) {
  const float* p = W + (size_t)(n0 + (lane & 15)) * K + k0 + (lane >> 4) * 8;
  v16bf b;
#pragma unroll
  for (int j = 0; j < 8; ++j) { b[j] = (__bf16)p[j]; b[j + 8] = (__bf16)p[16 + j]; }
  return b;
}

// C/D layout (ISA 7.12.2): VGPR r, lanes 0-15 -> (M=m0+r,    N=n0+lane)
//                                  lanes 16-31 -> (M=m0+r+8, N=n0+lane-16)
__device__ __forceinline__ void store_c_lds(float* Acc, int m0, int n0,
                                            int lane, v8f c) {
  const int hi = lane >> 4, lo = lane & 15;
#pragma unroll
  for (int r = 0; r < 8; ++r) Acc[(m0 + r + 8 * hi) * CS + n0 + lo] = c[r];
}

// Fused LayerNorm(+bias) + SiLU: Acc(f32) -> Ab(bf16). 4 threads per edge row;
// float4 LDS reads, cross-lane reduce over 4-lane groups, packed bf16 stores.
__device__ __forceinline__ void ln_silu(const float* __restrict__ Acc,
                                        __bf16* __restrict__ Ab,
                                        const float* __restrict__ bias,
                                        const float* __restrict__ gam,
                                        const float* __restrict__ bet,
                                        int tid) {
  const int e = tid >> 2;
  const int q = (tid & 3) * 32;
  const float4* rp = (const float4*)(Acc + e * CS + q);
  const float4* bp = (const float4*)(bias + q);
  float4 vbuf[8];
  float s = 0.f, s2 = 0.f;
#pragma unroll
  for (int i = 0; i < 8; ++i) {
    float4 v = rp[i];
    const float4 bb = bp[i];
    v.x += bb.x; v.y += bb.y; v.z += bb.z; v.w += bb.w;
    vbuf[i] = v;
    s  += (v.x + v.y) + (v.z + v.w);
    s2 += (v.x * v.x + v.y * v.y) + (v.z * v.z + v.w * v.w);
  }
  s  += __shfl_xor(s, 1, 32);  s  += __shfl_xor(s, 2, 32);
  s2 += __shfl_xor(s2, 1, 32); s2 += __shfl_xor(s2, 2, 32);
  const float mu   = s * (1.f / H);
  const float var  = fmaxf(s2 * (1.f / H) - mu * mu, 0.f);
  const float rstd = rsqrtf(var + 1e-5f);
  const float4* gp = (const float4*)(gam + q);
  const float4* ep = (const float4*)(bet + q);
  v2bf* op = (v2bf*)(Ab + e * AS + q);
#pragma unroll
  for (int i = 0; i < 8; ++i) {
    const float4 v = vbuf[i], g = gp[i], be = ep[i];
    float h0 = (v.x - mu) * rstd * g.x + be.x;
    float h1 = (v.y - mu) * rstd * g.y + be.y;
    float h2 = (v.z - mu) * rstd * g.z + be.z;
    float h3 = (v.w - mu) * rstd * g.w + be.w;
    h0 = h0 / (1.f + __expf(-h0));
    h1 = h1 / (1.f + __expf(-h1));
    h2 = h2 / (1.f + __expf(-h2));
    h3 = h3 / (1.f + __expf(-h3));
    v2bf p01 = { (__bf16)h0, (__bf16)h1 };
    v2bf p23 = { (__bf16)h2, (__bf16)h3 };
    op[i * 2]     = p01;
    op[i * 2 + 1] = p23;
  }
}

// ---------------------------------------------------------------------------
// One-time weight repack: f32 [Nout][K] -> fragment-packed bf16 B operands.
// One 32-lane wave per fragment; fully coalesced reads, 32B/lane packed write.
__global__ void __launch_bounds__(32)
pack_weights(const float* __restrict__ W1, const float* __restrict__ W2,
             const float* __restrict__ W3, __bf16* __restrict__ Wp) {
  const int ex   = blockIdx.x >> 6;
  const int fid  = blockIdx.x & (FRAGS_PER_EXPERT - 1);
  const int lane = threadIdx.x;
  const float* W; int K, n0, k0;
  if (fid < 16) {
    W = W1 + (size_t)ex * H * D_IN; K = D_IN;
    n0 = (fid >> 1) * 16; k0 = (fid & 1) * 32;
  } else if (fid < 48) {
    W = W2 + (size_t)ex * H * H; K = H;
    n0 = ((fid - 16) >> 2) * 16; k0 = ((fid - 16) & 3) * 32;
  } else {
    W = W3 + (size_t)ex * D_OUT * H; K = H;
    n0 = ((fid - 48) >> 2) * 16; k0 = ((fid - 48) & 3) * 32;
  }
  const float* p = W + (size_t)(n0 + (lane & 15)) * K + k0 + (lane >> 4) * 8;      // lane indexes N
  v16bf b;
#pragma unroll
  for (int j = 0; j < 8; ++j) { b[j] = (__bf16)p[j]; b[j + 8] = (__bf16)p[16 + j]; }
  *(v16bf*)(Wp + (size_t)blockIdx.x * FRAG_ELEMS + lane * 16) = b;
}

// ---------------------------------------------------------------------------
template<bool PACKED>
__global__ void __launch_bounds__(256)
radial_mlp_fused(const float* __restrict__ x,
                 const float* __restrict__ W1, const float* __restrict__ b1,
                 const float* __restrict__ g1, const float* __restrict__ be1,
                 const float* __restrict__ W2, const float* __restrict__ b2,
                 const float* __restrict__ g2, const float* __restrict__ be2,
                 const float* __restrict__ W3, const float* __restrict__ b3,
                 const __bf16* __restrict__ Wp,
                 float* __restrict__ out, int E) {
  __shared__ __bf16 Xt[TILE_E * XS];        //  9216 B
  __shared__ __bf16 Ab[TILE_E * AS];        // 17408 B
  __shared__ float  Acc[TILE_E * CS];       // 33792 B
  __shared__ float  Cst[6 * H + D_OUT];     //  3328 B  (total 63744 B)

  const int tid    = threadIdx.x;
  const int wave   = tid >> 5;
  const int lane   = tid & 31;
  const int expert = blockIdx.y;
  const int e0     = blockIdx.x * TILE_E;

  const float* W1e = W1 + (size_t)expert * H * D_IN;
  const float* W2e = W2 + (size_t)expert * H * H;
  const float* W3e = W3 + (size_t)expert * D_OUT * H;
  const int fbase  = expert * FRAGS_PER_EXPERT;

  // per-expert constants -> LDS
  for (int i = tid; i < 6 * H + D_OUT; i += 256) {
    float v;
    if      (i < 128) v = b1 [expert * H + i];
    else if (i < 256) v = g1 [expert * H + i - 128];
    else if (i < 384) v = be1[expert * H + i - 256];
    else if (i < 512) v = b2 [expert * H + i - 384];
    else if (i < 640) v = g2 [expert * H + i - 512];
    else if (i < 768) v = be2[expert * H + i - 640];
    else              v = b3 [expert * D_OUT + i - 768];
    Cst[i] = v;
  }
  // x tile -> bf16 LDS (zero-padded past E), paired loads/stores
  for (int i = tid; i < TILE_E * (D_IN / 2); i += 256) {
    const int e = i >> 5, c = (i & 31) * 2;
    float2 v = {0.f, 0.f};
    if (e0 + e < E) v = *(const float2*)(x + (size_t)(e0 + e) * D_IN + c);
    v2bf pkt = { (__bf16)v.x, (__bf16)v.y };
    *(v2bf*)(Xt + e * XS + c) = pkt;
  }
  __syncthreads();

  // ===== Layer 1: [64 x 64] x [64 x 128] -> Acc =====
  {
    const int n0 = wave * 16;
    const v16bf bf0 = PACKED ? load_frag(Wp, fbase + wave * 2 + 0, lane)
                             : load_b_gather<D_IN>(W1e, n0, 0, lane);
    const v16bf bf1 = PACKED ? load_frag(Wp, fbase + wave * 2 + 1, lane)
                             : load_b_gather<D_IN>(W1e, n0, 32, lane);
#pragma unroll
    for (int mt = 0; mt < 4; ++mt) {
      v8f c = {};
      v16bf a0 = load_a_lds(Xt, XS, mt * 16, 0, lane);
      c = __builtin_amdgcn_wmma_f32_16x16x32_bf16(false, a0, false, bf0, (short)0, c, false, false);
      v16bf a1 = load_a_lds(Xt, XS, mt * 16, 32, lane);
      c = __builtin_amdgcn_wmma_f32_16x16x32_bf16(false, a1, false, bf1, (short)0, c, false, false);
      store_c_lds(Acc, mt * 16, n0, lane, c);
    }
  }
  __syncthreads();
  ln_silu(Acc, Ab, Cst + 0, Cst + 128, Cst + 256, tid);
  __syncthreads();

  // ===== Layer 2: [64 x 128] x [128 x 128] -> Acc =====
  {
    const int n0 = wave * 16;
    v16bf bf[4];
#pragma unroll
    for (int ks = 0; ks < 4; ++ks)
      bf[ks] = PACKED ? load_frag(Wp, fbase + 16 + wave * 4 + ks, lane)
                      : load_b_gather<H>(W2e, n0, ks * 32, lane);
#pragma unroll
    for (int mt = 0; mt < 4; ++mt) {
      v8f c = {};
#pragma unroll
      for (int ks = 0; ks < 4; ++ks) {
        v16bf a = load_a_lds(Ab, AS, mt * 16, ks * 32, lane);
        c = __builtin_amdgcn_wmma_f32_16x16x32_bf16(false, a, false, bf[ks], (short)0, c, false, false);
      }
      store_c_lds(Acc, mt * 16, n0, lane, c);
    }
  }
  __syncthreads();
  ln_silu(Acc, Ab, Cst + 384, Cst + 512, Cst + 640, tid);
  __syncthreads();

  // ===== Layer 3: [64 x 128] x [128 x 64] -> global (+b3) =====
  {
    const int n0 = (wave >> 1) * 16;          // waves pair up per N-tile
    v16bf bf[4];
#pragma unroll
    for (int ks = 0; ks < 4; ++ks)
      bf[ks] = PACKED ? load_frag(Wp, fbase + 48 + (wave >> 1) * 4 + ks, lane)
                      : load_b_gather<H>(W3e, n0, ks * 32, lane);
    const int hi = lane >> 4, lo = lane & 15;
    const float bb = Cst[768 + n0 + lo];
#pragma unroll
    for (int t = 0; t < 2; ++t) {
      const int mt = (2 * wave + t) & 3;
      v8f c = {};
#pragma unroll
      for (int ks = 0; ks < 4; ++ks) {
        v16bf a = load_a_lds(Ab, AS, mt * 16, ks * 32, lane);
        c = __builtin_amdgcn_wmma_f32_16x16x32_bf16(false, a, false, bf[ks], (short)0, c, false, false);
      }
#pragma unroll
      for (int r = 0; r < 8; ++r) {
        const int e = e0 + mt * 16 + r + 8 * hi;
        if (e < E)
          out[((size_t)expert * E + e) * D_OUT + n0 + lo] = c[r] + bb;
      }
    }
  }
}

// ---------------------------------------------------------------------------
extern "C" void kernel_launch(void* const* d_in, const int* in_sizes, int n_in,
                              void* d_out, int out_size, void* d_ws, size_t ws_size,
                              hipStream_t stream) {
  const float* x   = (const float*)d_in[0];
  const float* W1  = (const float*)d_in[1];
  const float* b1  = (const float*)d_in[2];
  const float* g1  = (const float*)d_in[3];
  const float* be1 = (const float*)d_in[4];
  const float* W2  = (const float*)d_in[5];
  const float* b2  = (const float*)d_in[6];
  const float* g2  = (const float*)d_in[7];
  const float* be2 = (const float*)d_in[8];
  const float* W3  = (const float*)d_in[9];
  const float* b3  = (const float*)d_in[10];
  float* out = (float*)d_out;
  const int E = in_sizes[0] / D_IN;
  dim3 grid((E + TILE_E - 1) / TILE_E, N_EXP);

  if (ws_size >= WS_NEEDED && d_ws != nullptr) {
    __bf16* Wp = (__bf16*)d_ws;
    pack_weights<<<N_EXP * FRAGS_PER_EXPERT, 32, 0, stream>>>(W1, W2, W3, Wp);
    radial_mlp_fused<true><<<grid, 256, 0, stream>>>(
        x, W1, b1, g1, be1, W2, b2, g2, be2, W3, b3, Wp, out, E);
  } else {
    radial_mlp_fused<false><<<grid, 256, 0, stream>>>(
        x, W1, b1, g1, be1, W2, b2, g2, be2, W3, b3, nullptr, out, E);
  }
  (void)n_in; (void)out_size;
}
